// LocallyConnected_57698590655039
// MI455X (gfx1250) — matrix operand order repaired and checked by reading, not compiled, not run on hardware
//
#include <hip/hip_runtime.h>
#include <hip/hip_bf16.h>

typedef __attribute__((ext_vector_type(16))) __bf16 v16bf;
typedef __attribute__((ext_vector_type(8)))  __bf16 v8bf;
typedef __attribute__((ext_vector_type(8)))  float  v8f;

#define N_FEAT   8192
#define NGROUPS  512
#define GDIM     64
#define DDIM     128
#define BDIM     512
#define LDS_STRIDE 72   // padded col-major stride (bf16 elems) -> bank-friendly b128 loads

__global__ __launch_bounds__(256)
void lc_gemm_ln_kernel(const float* __restrict__ x,
                       const int*   __restrict__ idx,
                       const float* __restrict__ W,
                       const float* __restrict__ bias,
                       const float* __restrict__ gamma,
                       const float* __restrict__ beta,
                       float*       __restrict__ out)
{
    __shared__ __bf16 sWh[DDIM * LDS_STRIDE];   // W[n] hi, col-major [col][K]
    __shared__ __bf16 sWl[DDIM * LDS_STRIDE];   // W[n] lo
    __shared__ int    sIdx[GDIM];

    const int tid  = threadIdx.x;
    const int lane = tid & 31;
    const int wave = tid >> 5;
    const int lid  = lane & 15;    // M (A) / N (B,C,D) index
    const int half = lane >> 4;    // lane half selects K sub-range / M+8

    const int n     = blockIdx.x >> 2;                  // group: 4 blocks per group
    const int btile = ((blockIdx.x & 3) << 3) + wave;   // 32 batch tiles of 16 rows
    const int row0  = btile << 4;

    // ---- Stage W[n] (64x128 f32) into LDS as bf16 hi/lo, transposed to col-major ----
    const float* Wn = W + (unsigned)n * (GDIM * DDIM);
    #pragma unroll
    for (int j = 0; j < 32; ++j) {
        unsigned i   = (unsigned)(j * 256 + tid);  // coalesced over columns
        unsigned K   = i >> 7;
        unsigned col = i & 127u;
        float wv  = Wn[i];
        __bf16 hi = (__bf16)wv;
        sWh[col * LDS_STRIDE + K] = hi;
        sWl[col * LDS_STRIDE + K] = (__bf16)(wv - (float)hi);
    }
    if (tid < GDIM) sIdx[tid] = idx[n * GDIM + tid];
    __syncthreads();

    // ---- Gather A = x[row0+lid, idx[n,:]] ----
    // Uniform SGPR base + 32-bit unsigned element offset -> GVS addressing, 1 VALU/load.
    const unsigned rowoff = (unsigned)(row0 + lid) * (unsigned)N_FEAT;
    float xg[2][16];
    #pragma unroll
    for (int c = 0; c < 2; ++c) {
        #pragma unroll
        for (int e = 0; e < 16; ++e) {
            int K = 32 * c + 16 * half + e;
            xg[c][e] = x[rowoff + (unsigned)sIdx[K]];
        }
    }
    // hi/lo bf16 split (A 16x32 in this code's contiguous k order: lanes 0-15 hold K = e(0..15); lanes 16-31: +16)
    v16bf ah[2], al[2];
    #pragma unroll
    for (int c = 0; c < 2; ++c) {
        #pragma unroll
        for (int e = 0; e < 16; ++e) {
            float xv  = xg[c][e];
            __bf16 hi = (__bf16)xv;
            ah[c][e]  = hi;
            al[c][e]  = (__bf16)(xv - (float)hi);
        }
    }

    // ---- 8 N-subtiles of 16; bf16x3 WMMA accumulation in f32 ----
    v8f accs[8];
    #pragma unroll
    for (int t = 0; t < 8; ++t) {
        const int col = t * 16 + lid;
        const __bf16* ph = &sWh[col * LDS_STRIDE];
        const __bf16* pl = &sWl[col * LDS_STRIDE];
        v8f acc = {};
        #pragma unroll
        for (int c = 0; c < 2; ++c) {
            // B 32x16 in the same contiguous k order as A: lane holds K = 16*half + (0..15), contiguous
            const int k0 = 32 * c + 16 * half;
            v8bf h0 = *(const v8bf*)(ph + k0);
            v8bf h1 = *(const v8bf*)(ph + k0 + 8);
            v8bf l0 = *(const v8bf*)(pl + k0);
            v8bf l1 = *(const v8bf*)(pl + k0 + 8);
            v16bf bh = __builtin_shufflevector(h0, h1, 0,1,2,3,4,5,6,7,8,9,10,11,12,13,14,15);
            v16bf bl = __builtin_shufflevector(l0, l1, 0,1,2,3,4,5,6,7,8,9,10,11,12,13,14,15);
            acc = __builtin_amdgcn_wmma_f32_16x16x32_bf16(false, ah[c], false, bh,
                                                          (short)0, acc, false, false);
            acc = __builtin_amdgcn_wmma_f32_16x16x32_bf16(false, al[c], false, bh,
                                                          (short)0, acc, false, false);
            acc = __builtin_amdgcn_wmma_f32_16x16x32_bf16(false, ah[c], false, bl,
                                                          (short)0, acc, false, false);
        }
        accs[t] = acc;
    }

    // ---- Fused bias + LeakyReLU + LayerNorm(D=128) epilogue ----
    float bvals[8], gvals[8], betv[8];
    #pragma unroll
    for (int t = 0; t < 8; ++t) {
        unsigned col = (unsigned)(t * 16 + lid);
        bvals[t] = bias[(unsigned)n * DDIM + col];
        gvals[t] = gamma[col];
        betv[t]  = beta[col];
    }

    // One 32-bit per-lane base; v/t offsets fold into the 24-bit store immediates.
    float* outp = out + ((unsigned)(row0 + 8 * half) * (unsigned)(NGROUPS * DDIM)
                         + (unsigned)n * DDIM + (unsigned)lid);

    #pragma unroll
    for (int v = 0; v < 8; ++v) {        // row M = v + 8*half
        float h[8];
        float s1 = 0.f, s2 = 0.f;
        #pragma unroll
        for (int t = 0; t < 8; ++t) {
            float z = accs[t][v] + bvals[t];
            z = (z >= 0.f) ? z : 0.2f * z;
            h[t] = z;
            s1 += z;
            s2 += z * z;
        }
        // reduce across the 16 lanes of this half (N dimension)
        #pragma unroll
        for (int m = 1; m < 16; m <<= 1) {
            s1 += __shfl_xor(s1, m, 32);
            s2 += __shfl_xor(s2, m, 32);
        }
        float mu  = s1 * (1.0f / 128.0f);
        float var = fmaf(-mu, mu, s2 * (1.0f / 128.0f));
        float rs  = rsqrtf(var + 1e-5f);
        #pragma unroll
        for (int t = 0; t < 8; ++t) {
            float o = fmaf(h[t] - mu, rs * gvals[t], betv[t]);
            // write-once stream: non-temporal so the 134MB output doesn't evict x/W from L2
            __builtin_nontemporal_store(o, outp + (unsigned)(v * (NGROUPS * DDIM) + t * 16));
        }
    }
}

extern "C" void kernel_launch(void* const* d_in, const int* in_sizes, int n_in,
                              void* d_out, int out_size, void* d_ws, size_t ws_size,
                              hipStream_t stream) {
    (void)in_sizes; (void)n_in; (void)out_size; (void)d_ws; (void)ws_size;
    const float* x     = (const float*)d_in[0];
    const int*   idx   = (const int*)  d_in[1];
    const float* W     = (const float*)d_in[2];
    const float* b     = (const float*)d_in[3];
    const float* gamma = (const float*)d_in[4];
    const float* beta  = (const float*)d_in[5];
    float*       out   = (float*)d_out;

    // 512 groups x 4 blocks; each block = 8 waves = 8 batch-tiles of 16 rows
    dim3 grid(NGROUPS * 4);
    dim3 block(256);
    lc_gemm_ln_kernel<<<grid, block, 0, stream>>>(x, idx, W, b, gamma, beta, out);
}
